// OrderedNeuronLSTMLayer_85358180041495
// MI455X (gfx1250) — matrix operand order repaired and checked by reading, not compiled, not run on hardware
//
#include <hip/hip_runtime.h>
#include <hip/hip_bf16.h>

// ---------------------------------------------------------------------------
// ON-LSTM layer for MI455X (gfx1250, wave32, WMMA) — persistent-kernel design.
// S=512, B=64, E=512, H=512.  per step: 6 gates = [x_t|h] @ W_g + b_g,
// mf/mi gates -> cumsum(softmax(.)), then ON-LSTM cell update.
//
//   kernel 1 (wconv): 6 weight mats f32[1024,512] -> bf16 transposed [n][k];
//                     zero-inits the grid-barrier counter.
//   kernel 2 (persistent, ONE launch): 32 blocks x 256 threads. Each block
//     owns 96 of the 3072 fused gate columns; its bf16 weight slice is DMA'd
//     into LDS once via the Tensor Data Mover (tensor_load_to_lds +
//     s_wait_tensorcnt) and stays resident for all 512 steps. Steps are
//     separated by an in-kernel grid barrier (atomic counter + s_sleep).
// ---------------------------------------------------------------------------

typedef __bf16 bf16_t;
typedef __attribute__((ext_vector_type(16))) __bf16 v16bf;
typedef __attribute__((ext_vector_type(8)))  __bf16 v8bf;
typedef __attribute__((ext_vector_type(8)))  float  v8f;
typedef __attribute__((ext_vector_type(4)))  unsigned int v4u;
typedef __attribute__((ext_vector_type(8)))  int v8i;
typedef __attribute__((ext_vector_type(4)))  int v4i;

#define S_LEN   512
#define BATCH   64
#define EDIM    512
#define HDIM    512
#define KDIM    (EDIM + HDIM)          // 1024
#define NGATES  6
#define NTOT    (NGATES * HDIM)        // 3072 fused gate columns

#define NBLOCKS 32
#define PTHREADS 256                   // 8 waves
#define COLS_PER_BLOCK (NTOT / NBLOCKS)          // 96 -> 6 N-tiles
#define NTILES_PER_WAVE 3                        // 8 waves: 4 M-tiles x 2 N-halves
#define SMEM_W_BYTES (COLS_PER_BLOCK * KDIM * 2) // 192 KB bf16 weight slice

// ws layout (bytes)
#define WS_WT_OFF  0
#define WS_WT_SZ   ((size_t)NTOT * KDIM * 2)     // 6.29 MB bf16 weights
#define WS_G_OFF   (WS_WT_OFF + WS_WT_SZ)
#define WS_G_SZ    ((size_t)BATCH * NTOT * 4)    // 0.79 MB gate pre-activations
#define WS_H_OFF   (WS_G_OFF + WS_G_SZ)
#define WS_H_SZ    ((size_t)BATCH * HDIM * 4)
#define WS_C_OFF   (WS_H_OFF + WS_H_SZ)
#define WS_C_SZ    ((size_t)BATCH * HDIM * 4)
#define WS_CNT_OFF (WS_C_OFF + WS_C_SZ)          // barrier counter (128B slot)

__device__ __forceinline__ float sigmoidf_fast(float x) {
  return 1.0f / (1.0f + __expf(-x));
}

// Monotonic grid-wide barrier: every block arrives once; all proceed when the
// counter reaches `target`. Fences give agent-scope visibility of prior stores.
__device__ __forceinline__ void grid_barrier(unsigned* cnt, unsigned target) {
  __threadfence();            // make this block's global stores agent-visible
  __syncthreads();
  if (threadIdx.x == 0) {
    __hip_atomic_fetch_add(cnt, 1u, __ATOMIC_ACQ_REL, __HIP_MEMORY_SCOPE_AGENT);
    while (__hip_atomic_load(cnt, __ATOMIC_ACQUIRE, __HIP_MEMORY_SCOPE_AGENT) < target)
      __builtin_amdgcn_s_sleep(2);
  }
  __syncthreads();
  __threadfence();            // acquire: invalidate stale near caches
}

// Build one WMMA A-fragment (16-bit A 16x32 ISA layout) from 32 consecutive
// f32 K-values of one row: elems 0..7 = K kb..kb+7, elems 8..15 = kb+16..+23.
__device__ __forceinline__ v16bf make_afrag(const float* __restrict__ srcp) {
  float4 fa = *(const float4*)(srcp);
  float4 fb = *(const float4*)(srcp + 4);
  float4 fc = *(const float4*)(srcp + 16);
  float4 fd = *(const float4*)(srcp + 20);
  v16bf a;
  a[0]  = (bf16_t)fa.x; a[1]  = (bf16_t)fa.y; a[2]  = (bf16_t)fa.z; a[3]  = (bf16_t)fa.w;
  a[4]  = (bf16_t)fb.x; a[5]  = (bf16_t)fb.y; a[6]  = (bf16_t)fb.z; a[7]  = (bf16_t)fb.w;
  a[8]  = (bf16_t)fc.x; a[9]  = (bf16_t)fc.y; a[10] = (bf16_t)fc.z; a[11] = (bf16_t)fc.w;
  a[12] = (bf16_t)fd.x; a[13] = (bf16_t)fd.y; a[14] = (bf16_t)fd.z; a[15] = (bf16_t)fd.w;
  return a;
}

// ---------------------------------------------------------------------------
// Weight conversion: wt[(g*H + n)*K + k] = bf16( W_g[k*H + n] ); init barrier.
// ---------------------------------------------------------------------------
__global__ void wconv_kernel(const float* __restrict__ W0, const float* __restrict__ W1,
                             const float* __restrict__ W2, const float* __restrict__ W3,
                             const float* __restrict__ W4, const float* __restrict__ W5,
                             bf16_t* __restrict__ wt, unsigned* __restrict__ cnt) {
  if (blockIdx.x == 0 && threadIdx.x == 0) *cnt = 0u;   // barrier counter init
  size_t idx = (size_t)blockIdx.x * blockDim.x + threadIdx.x;
  if (idx >= (size_t)NGATES * HDIM * KDIM) return;
  int g   = (int)(idx / ((size_t)HDIM * KDIM));
  int rem = (int)(idx % ((size_t)HDIM * KDIM));
  int n = rem / KDIM;
  int k = rem % KDIM;
  const float* Ws[NGATES] = {W0, W1, W2, W3, W4, W5};
  wt[idx] = (bf16_t)Ws[g][(size_t)k * HDIM + n];
}

// ---------------------------------------------------------------------------
// Persistent ON-LSTM scan. grid = 32 blocks x 256 threads, LDS = 192 KB.
// ---------------------------------------------------------------------------
__global__ __launch_bounds__(PTHREADS, 1)
void onlstm_persistent(const float* __restrict__ x,        // [S,B,E]
                       const float* __restrict__ h0,       // [B,H]
                       const float* __restrict__ c0,       // [B,H]
                       const bf16_t* __restrict__ wt,      // [NTOT][KDIM] bf16
                       const float* __restrict__ bias_f, const float* __restrict__ bias_i,
                       const float* __restrict__ bias_o, const float* __restrict__ bias_c,
                       const float* __restrict__ bias_mf, const float* __restrict__ bias_mi,
                       const int*   __restrict__ seqlen,   // [B]
                       float* __restrict__ gws,            // [B][NTOT] f32 scratch
                       float* __restrict__ h_cur, float* __restrict__ c_cur,
                       unsigned* __restrict__ cnt,
                       float* __restrict__ out_h,          // [S,B,H]
                       float* __restrict__ out_sh,         // [B,H]
                       float* __restrict__ out_sc) {       // [B,H]
  extern __shared__ char smem[];
  bf16_t* wlds = (bf16_t*)smem;          // [96][1024] bf16 weight slice

  const int tid     = threadIdx.x;
  const int lane    = tid & 31;
  const int wave    = tid >> 5;
  const int blk     = blockIdx.x;
  const int halfsel = lane >> 4;         // 0: lanes 0-15, 1: lanes 16-31
  const int l15     = lane & 15;

  // ---- one-time: DMA this block's bf16 weight slice into LDS via TDM ------
#if __has_builtin(__builtin_amdgcn_tensor_load_to_lds)
  if (wave == 0) {
    // D# (CDNA5 ISA): 2D tensor, data_size=2B,
    // tile_dim0=KDIM elements per row, tile_dim1=COLS_PER_BLOCK rows,
    // contiguous (tensor_dim0_stride = KDIM), dest = dynamic-LDS base (0).
    const unsigned long long ga =
        (unsigned long long)(const void*)(wt + (size_t)blk * COLS_PER_BLOCK * KDIM);
    v4u g0;
    g0[0] = 1u;                                        // count=1, user mode
    g0[1] = 0u;                                        // lds_addr (bytes)
    g0[2] = (unsigned)(ga & 0xFFFFFFFFu);              // global_addr[31:0]
    g0[3] = (unsigned)((ga >> 32) & 0x1FFFFFFu)        // global_addr[56:32]
          | (2u << 30);                                // type=2 ("image")
    v8i g1;
    g1[0] = (int)(1u << 16);                           // data_size=1 -> 2B, mask=0
    g1[1] = (int)((KDIM & 0xFFFFu) << 16);             // tensor_dim0[15:0]
    g1[2] = (int)((KDIM >> 16) | ((COLS_PER_BLOCK & 0xFFFFu) << 16)); // td0 hi | td1 lo
    g1[3] = (int)((COLS_PER_BLOCK >> 16) | ((KDIM & 0xFFFFu) << 16)); // td1 hi | tile_dim0
    g1[4] = (int)(COLS_PER_BLOCK & 0xFFFFu);           // tile_dim1 | tile_dim2=0
    g1[5] = (int)KDIM;                                 // tensor_dim0_stride[31:0]
    g1[6] = 0;                                         // stride hi | dim1_stride lo
    g1[7] = 0;
    v4i zz4 = (v4i){0, 0, 0, 0};
    v8i zz8 = (v8i){0, 0, 0, 0, 0, 0, 0, 0};
    // 6-arg form (clang-23 / therock headers): (g0, g1, g2, g3, g4, cpol)
    __builtin_amdgcn_tensor_load_to_lds(g0, g1, zz4, zz4, zz8, 0);
    __builtin_amdgcn_s_wait_tensorcnt(0);
  }
#else
  {
    const uint4* s4 = (const uint4*)(wt + (size_t)blk * COLS_PER_BLOCK * KDIM);
    uint4* d4 = (uint4*)wlds;
    const int nchunk = COLS_PER_BLOCK * KDIM * 2 / 16;
    for (int i = tid; i < nchunk; i += PTHREADS) d4[i] = s4[i];
  }
#endif
  __syncthreads();

  // GEMM role: wave -> (M-tile, N-half).  M-tiles: 4 x 16 batch rows;
  // N-halves: 2 x 48 columns (3 x 16-col WMMA tiles each).
  const int mt   = wave & 3;
  const int nh   = wave >> 2;
  const int arow = mt * 16 + l15;        // batch row this lane supplies for A

  // loop-invariant per-lane bias values (one per owned N-tile)
  float bvv[NTILES_PER_WAVE];
  {
    const float* bptr[NGATES] = {bias_f, bias_i, bias_o, bias_c, bias_mf, bias_mi};
#pragma unroll
    for (int tI = 0; tI < NTILES_PER_WAVE; ++tI) {
      const int ng = blk * COLS_PER_BLOCK + nh * 48 + tI * 16 + l15;
      bvv[tI] = bptr[ng >> 9][ng & (HDIM - 1)];
    }
  }

  unsigned target = 0;

  for (int t = 0; t < S_LEN; ++t) {
    const float* hsrc = (t == 0) ? h0 : h_cur;

    // ================= phase 1: fused 6-gate GEMM slice ====================
    v8f acc[NTILES_PER_WAVE];
#pragma unroll
    for (int i = 0; i < NTILES_PER_WAVE; ++i)
      acc[i] = (v8f){0.f, 0.f, 0.f, 0.f, 0.f, 0.f, 0.f, 0.f};

    const float* xrow = x + ((size_t)t * BATCH + arow) * EDIM;
    const float* hrow = hsrc + (size_t)arow * HDIM;
    if (t + 1 < S_LEN)                     // warm L2 for next step's x row
      __builtin_prefetch(xrow + (size_t)BATCH * EDIM, 0, 1);

    // K-loop split at the x|h seam (k=512): no per-iteration source select.
#pragma unroll 2
    for (int k0 = 0; k0 < EDIM; k0 += 32) {
      v16bf afrag = make_afrag(xrow + k0 + halfsel * 8);
      const int bk = k0 + halfsel * 8;
      v16bf bfr[NTILES_PER_WAVE];
#pragma unroll
      for (int tI = 0; tI < NTILES_PER_WAVE; ++tI) {
        const bf16_t* bp = wlds + (size_t)(nh * 48 + tI * 16 + l15) * KDIM + bk;
        bfr[tI] = __builtin_shufflevector(*(const v8bf*)bp, *(const v8bf*)(bp + 16),
            0, 1, 2, 3, 4, 5, 6, 7, 8, 9, 10, 11, 12, 13, 14, 15);
      }
#pragma unroll
      for (int tI = 0; tI < NTILES_PER_WAVE; ++tI)
        acc[tI] = __builtin_amdgcn_wmma_f32_16x16x32_bf16(
            false, afrag, false, bfr[tI], (short)0, acc[tI], false, false);
    }
#pragma unroll 2
    for (int k0 = 0; k0 < HDIM; k0 += 32) {
      v16bf afrag = make_afrag(hrow + k0 + halfsel * 8);
      const int bk = EDIM + k0 + halfsel * 8;
      v16bf bfr[NTILES_PER_WAVE];
#pragma unroll
      for (int tI = 0; tI < NTILES_PER_WAVE; ++tI) {
        const bf16_t* bp = wlds + (size_t)(nh * 48 + tI * 16 + l15) * KDIM + bk;
        bfr[tI] = __builtin_shufflevector(*(const v8bf*)bp, *(const v8bf*)(bp + 16),
            0, 1, 2, 3, 4, 5, 6, 7, 8, 9, 10, 11, 12, 13, 14, 15);
      }
#pragma unroll
      for (int tI = 0; tI < NTILES_PER_WAVE; ++tI)
        acc[tI] = __builtin_amdgcn_wmma_f32_16x16x32_bf16(
            false, afrag, false, bfr[tI], (short)0, acc[tI], false, false);
    }

    // spill accumulators (+bias) to global gate scratch.
    // C/D layout: VGPR j -> (M = mt*16 + j + 8*half, N = lane&15)
#pragma unroll
    for (int tI = 0; tI < NTILES_PER_WAVE; ++tI) {
      const int ng = blk * COLS_PER_BLOCK + nh * 48 + tI * 16 + l15;
#pragma unroll
      for (int j = 0; j < 8; ++j) {
        const int m = mt * 16 + j + halfsel * 8;
        gws[(size_t)m * NTOT + ng] = acc[tI][j] + bvv[tI];
      }
    }

    target += NBLOCKS;
    grid_barrier(cnt, target);

    // ================= phase 2: gate math (2 batch rows / block) ===========
    if (wave < 2) {
      const int b = blk * 2 + wave;
      const float* grow = gws + (size_t)b * NTOT;
      const int nbase = lane * 16;                    // 16 contiguous columns

      float cmf[16], cmi[16];
      // cumax(g) = cumsum(softmax(g)) along H for master gates
#pragma unroll
      for (int which = 0; which < 2; ++which) {
        const float* lp = grow + (4 + which) * HDIM;
        float e[16];
        float mx = -3.402823466e38f;
#pragma unroll
        for (int j = 0; j < 16; ++j) { e[j] = lp[nbase + j]; mx = fmaxf(mx, e[j]); }
#pragma unroll
        for (int m = 16; m >= 1; m >>= 1) mx = fmaxf(mx, __shfl_xor(mx, m, 32));
        float run = 0.f;
#pragma unroll
        for (int j = 0; j < 16; ++j) { float ev = __expf(e[j] - mx); run += ev; e[j] = run; }
        float scan = run;                             // wave32 inclusive scan
#pragma unroll
        for (int d = 1; d < 32; d <<= 1) {
          float nb = __shfl_up(scan, d, 32);
          if (lane >= d) scan += nb;
        }
        float excl = scan - run;
        float inv  = 1.0f / __shfl(scan, 31, 32);
        float* dst = which ? cmi : cmf;
#pragma unroll
        for (int j = 0; j < 16; ++j) dst[j] = (excl + e[j]) * inv;
      }

      const int slast = seqlen[b] - 1;
      const float* csrc = (t == 0) ? c0 : c_cur;
#pragma unroll
      for (int j = 0; j < 16; ++j) {
        const int n = nbase + j;
        float ft = sigmoidf_fast(grow[0 * HDIM + n]);
        float it = sigmoidf_fast(grow[1 * HDIM + n]);
        float ot = sigmoidf_fast(grow[2 * HDIM + n]);
        float ch = tanhf(grow[3 * HDIM + n]);
        float mf = cmf[j], mi = cmi[j];
        float w  = mf * mi;
        float fh = ft * w + (mf - w);
        float ih = it * w + (mi - w);
        float cn = fh * csrc[(size_t)b * HDIM + n] + ih * ch;
        float hn = ot * cn;
        h_cur[(size_t)b * HDIM + n] = hn;
        c_cur[(size_t)b * HDIM + n] = cn;
        out_h[((size_t)t * BATCH + b) * HDIM + n] = hn;
        if (slast == t) {
          out_sh[(size_t)b * HDIM + n] = hn;
          out_sc[(size_t)b * HDIM + n] = cn;
        }
      }
    }

    target += NBLOCKS;
    grid_barrier(cnt, target);   // h/c visible to next step's GEMM
  }
}

// ---------------------------------------------------------------------------
extern "C" void kernel_launch(void* const* d_in, const int* in_sizes, int n_in,
                              void* d_out, int out_size, void* d_ws, size_t ws_size,
                              hipStream_t stream) {
  const float* x   = (const float*)d_in[0];
  const float* h0  = (const float*)d_in[1];
  const float* c0  = (const float*)d_in[2];
  const float* Wf  = (const float*)d_in[3];  const float* bF  = (const float*)d_in[4];
  const float* Wi  = (const float*)d_in[5];  const float* bI  = (const float*)d_in[6];
  const float* Wo  = (const float*)d_in[7];  const float* bO  = (const float*)d_in[8];
  const float* Wc  = (const float*)d_in[9];  const float* bC  = (const float*)d_in[10];
  const float* Wmf = (const float*)d_in[11]; const float* bMF = (const float*)d_in[12];
  const float* Wmi = (const float*)d_in[13]; const float* bMI = (const float*)d_in[14];
  const int*   seql = (const int*)d_in[15];

  bf16_t*   wt    = (bf16_t*)  ((char*)d_ws + WS_WT_OFF);
  float*    gws   = (float*)   ((char*)d_ws + WS_G_OFF);
  float*    h_cur = (float*)   ((char*)d_ws + WS_H_OFF);
  float*    c_cur = (float*)   ((char*)d_ws + WS_C_OFF);
  unsigned* cnt   = (unsigned*)((char*)d_ws + WS_CNT_OFF);

  float* out_h  = (float*)d_out;
  float* out_sh = out_h + (size_t)S_LEN * BATCH * HDIM;
  float* out_sc = out_sh + (size_t)BATCH * HDIM;

  // 1) weights -> bf16 [n][k]; zero the grid-barrier counter (stream-ordered)
  {
    int total  = NGATES * HDIM * KDIM;
    int blocks = (total + 255) / 256;
    wconv_kernel<<<blocks, 256, 0, stream>>>(Wf, Wi, Wo, Wc, Wmf, Wmi, wt, cnt);
  }

  // 2) the whole 512-step recurrence: ONE persistent kernel
  onlstm_persistent<<<NBLOCKS, PTHREADS, SMEM_W_BYTES, stream>>>(
      x, h0, c0, wt, bF, bI, bO, bC, bMF, bMI, seql,
      gws, h_cur, c_cur, cnt, out_h, out_sh, out_sc);
}
